// GCN_18726057411085
// MI455X (gfx1250) — compile-verified, matrix operand order checked by reading, not run on hardware
//
#include <hip/hip_runtime.h>
#include <math.h>

typedef __attribute__((ext_vector_type(16))) _Float16 v16h;
typedef __attribute__((ext_vector_type(8)))  _Float16 v8h;
typedef __attribute__((ext_vector_type(8)))  float    v8f;

// ---------------------------------------------------------------------------
// degree / normalization
// ---------------------------------------------------------------------------
__global__ void deg_init_kernel(float* deg, int N) {
    int i = blockIdx.x * blockDim.x + threadIdx.x;
    if (i < N) deg[i] = 1.0f;                      // self-loop contributes 1
}

__global__ void deg_edges_kernel(const int* dst, float* deg, int E) {
    int e = blockIdx.x * blockDim.x + threadIdx.x;
    if (e < E) atomicAdd(&deg[dst[e]], 1.0f);
}

__global__ void rsqrt_kernel(float* deg_dis, int N) {
    int i = blockIdx.x * blockDim.x + threadIdx.x;
    if (i < N) deg_dis[i] = rsqrtf(deg_dis[i]);    // deg >= 1 always
}

// ---------------------------------------------------------------------------
// x (N x Kreal fp32) -> xh (N x Kpad f16, zero-padded). grid = (Kpad/256, N)
// ---------------------------------------------------------------------------
__global__ void cvt_f16_pad_kernel(const float* __restrict__ x, _Float16* __restrict__ xh,
                                   int Kreal, int Kpad) {
    int k    = blockIdx.x * blockDim.x + threadIdx.x;
    int node = blockIdx.y;
    if (k >= Kpad) return;
    float v = (k < Kreal) ? x[(size_t)node * Kreal + k] : 0.0f;
    xh[(size_t)node * Kpad + k] = (_Float16)v;
}

// ---------------------------------------------------------------------------
// Pack W (Kreal x 128, fp32 row-major) into per-lane WMMA B fragments (f16).
// Fragment layout (CDNA5 ISA, 16-bit 32x16 B):
//   lane l (0..31): column n = 16*nt + (l&15)
//   j<8  -> k = 32*kt + j      + (l<16 ? 0 : 8)
//   j>=8 -> k = 32*kt + j + 8  + (l<16 ? 0 : 8)
// out index = ((kt*8 + nt)*32 + l)*16 + j
// ---------------------------------------------------------------------------
__global__ void pack_w_kernel(const float* W, _Float16* out, int Kreal, int Kpad) {
    int t = blockIdx.x * blockDim.x + threadIdx.x;
    int total = Kpad * 128;
    if (t >= total) return;
    int j  =  t        & 15;
    int l  = (t >> 4)  & 31;
    int nt = (t >> 9)  & 7;
    int kt =  t >> 12;
    int n  = nt * 16 + (l & 15);
    int k  = kt * 32 + ((j < 8) ? j : j + 8) + ((l < 16) ? 0 : 8);
    float v = (k < Kreal) ? W[k * 128 + n] : 0.0f;
    out[t] = (_Float16)v;
}

// ---------------------------------------------------------------------------
// WMMA GEMM: C[M x 128] = A[M x Kpad](f16, row-major) @ Bpk   (f32 accumulate)
// block = 256 threads = 8 waves; block -> 16 rows x 128 cols; wave -> 16 cols.
// A tile (16 x 32 f16 = 1KB) staged via async DMA into double-buffered LDS:
//   threads 0..63 each issue one global_load_async_to_lds_b128 (16B chunk),
//   pipelined one tile deep, synchronized with s_wait_asynccnt + barrier.
// M must be a multiple of 16 (50000 = 16 * 3125); Kpad a multiple of 32.
// ---------------------------------------------------------------------------
__global__ void __launch_bounds__(256)
gemm_wmma_f16_kernel(const _Float16* __restrict__ A, int lda, int ksteps,
                     const _Float16* __restrict__ Bpk, float* __restrict__ C) {
    __shared__ __align__(16) _Float16 Atile[2][16 * 32];

    const int tid  = threadIdx.x;
    const int wave = tid >> 5;
    const int lane = tid & 31;
    const int m0   = blockIdx.x * 16;

    // async-copy assignment (threads 0..63): 16 rows x 4 chunks of 16B
    const int cprow = tid >> 2;                        // row 0..15
    const int cpchk = tid & 3;                         // 16B chunk in row
    const _Float16* gbase = A + (size_t)(m0 + cprow) * lda + cpchk * 8;
    // flat address of LDS object: low 32 bits == LDS byte offset (aperture rule)
    unsigned ldsDst[2];
    ldsDst[0] = (unsigned)(size_t)&Atile[0][cprow * 32 + cpchk * 8];
    ldsDst[1] = (unsigned)(size_t)&Atile[1][cprow * 32 + cpchk * 8];

    auto issue_tile = [&](int kt) {
        unsigned long long ga = (unsigned long long)(size_t)(gbase + (size_t)kt * 32);
        asm volatile("global_load_async_to_lds_b128 %0, %1, off"
                     :: "v"(ldsDst[kt & 1]), "v"(ga) : "memory");
    };

    if (tid < 64) issue_tile(0);

    v8f acc = {};
    for (int kt = 0; kt < ksteps; ++kt) {
        if (kt + 1 < ksteps) {
            if (tid < 64) issue_tile(kt + 1);
            asm volatile("s_wait_asynccnt 0x1" ::: "memory");  // tile kt landed
        } else {
            asm volatile("s_wait_asynccnt 0x0" ::: "memory");
        }
        __syncthreads();

        // A fragment from LDS: lane holds row (lane&15); halves per ISA layout
        const _Float16* ap = &Atile[kt & 1][(lane & 15) * 32 + ((lane < 16) ? 0 : 8)];
        v8h a_lo = *(const v8h*)ap;          // K 0..7   (or 8..15)
        v8h a_hi = *(const v8h*)(ap + 16);   // K 16..23 (or 24..31)

        // B fragment: pre-packed, 16 contiguous f16 per lane (two 16B loads)
        const _Float16* bp = Bpk + (((size_t)kt * 8 + wave) * 32 + lane) * 16;
        v8h b_lo = *(const v8h*)bp;
        v8h b_hi = *(const v8h*)(bp + 8);

        v16h a = __builtin_shufflevector(a_lo, a_hi, 0,1,2,3,4,5,6,7,8,9,10,11,12,13,14,15);
        v16h b = __builtin_shufflevector(b_lo, b_hi, 0,1,2,3,4,5,6,7,8,9,10,11,12,13,14,15);

        acc = __builtin_amdgcn_wmma_f32_16x16x32_f16(
                  /*neg_a=*/false, a, /*neg_b=*/false, b,
                  /*c_mod=*/(short)0, acc, /*reuse_a=*/false, /*reuse_b=*/false);

        __syncthreads();                      // tile buffer free for kt+2
    }

    // C/D layout: VGPR r -> row (lane<16 ? r : 8+r), col = lane&15
    const int col   = lane & 15;
    const int rbase = (lane < 16) ? 0 : 8;
    float* cp = C + (size_t)m0 * 128 + wave * 16 + col;
#pragma unroll
    for (int r = 0; r < 8; ++r)
        cp[(size_t)(rbase + r) * 128] = acc[r];
}

// ---------------------------------------------------------------------------
// aggregation: agg = dis[i]^2 * h[i]  (self loop), then atomic scatter of edges
// ---------------------------------------------------------------------------
__global__ void agg_init_kernel(const float* h, const float* dis, float* agg, int N) {
    int t = blockIdx.x * blockDim.x + threadIdx.x;
    if (t >= N * 32) return;
    int node = t >> 5;
    int c    = (t & 31) * 4;
    float s  = dis[node]; s = s * s;
    const float4 v = *(const float4*)(h + (size_t)node * 128 + c);
    float4 o; o.x = v.x * s; o.y = v.y * s; o.z = v.z * s; o.w = v.w * s;
    *(float4*)(agg + (size_t)node * 128 + c) = o;
}

__global__ void scatter_kernel(const int* src, const int* dst, const float* dis,
                               const float* h, float* agg, int E) {
    int t = blockIdx.x * blockDim.x + threadIdx.x;
    int e = t >> 5;
    if (e >= E) return;
    int c = (t & 31) * 4;
    int s = src[e], d = dst[e];
    float nrm = dis[s] * dis[d];
    const float4 v = *(const float4*)(h + (size_t)s * 128 + c);
    float* ap = agg + (size_t)d * 128 + c;
    atomicAdd(ap + 0, v.x * nrm);
    atomicAdd(ap + 1, v.y * nrm);
    atomicAdd(ap + 2, v.z * nrm);
    atomicAdd(ap + 3, v.w * nrm);
}

__global__ void bias_relu_f16_kernel(const float* agg, const float* b,
                                     _Float16* out, int total) {
    int t = blockIdx.x * blockDim.x + threadIdx.x;
    if (t >= total) return;
    float v = agg[t] + b[t & 127];
    out[t] = (_Float16)fmaxf(v, 0.0f);
}

// ---------------------------------------------------------------------------
// output layer: logits = h @ Wo + bo, softmax over C=10. One wave per node.
// ---------------------------------------------------------------------------
__global__ void __launch_bounds__(256)
out_softmax_kernel(const _Float16* h, const float* Wo, const float* bo,
                   float* out, int N) {
    int node = blockIdx.x * 8 + (threadIdx.x >> 5);
    int lane = threadIdx.x & 31;
    if (node >= N) return;

    float hv[4];
    const _Float16* hp = h + (size_t)node * 128 + lane * 4;
#pragma unroll
    for (int j = 0; j < 4; ++j) hv[j] = (float)hp[j];

    float s[10];
#pragma unroll
    for (int c = 0; c < 10; ++c) {
        float p = 0.0f;
#pragma unroll
        for (int j = 0; j < 4; ++j)
            p += hv[j] * Wo[(lane * 4 + j) * 10 + c];
#pragma unroll
        for (int off = 16; off >= 1; off >>= 1)
            p += __shfl_xor(p, off, 32);
        s[c] = p + bo[c];
    }
    float m = s[0];
#pragma unroll
    for (int c = 1; c < 10; ++c) m = fmaxf(m, s[c]);
    float e[10], z = 0.0f;
#pragma unroll
    for (int c = 0; c < 10; ++c) { e[c] = __expf(s[c] - m); z += e[c]; }
    float rz = 1.0f / z;
    if (lane < 10) out[(size_t)node * 10 + lane] = e[lane] * rz;
}

// ---------------------------------------------------------------------------
extern "C" void kernel_launch(void* const* d_in, const int* in_sizes, int n_in,
                              void* d_out, int out_size, void* d_ws, size_t ws_size,
                              hipStream_t stream) {
    const float* x   = (const float*)d_in[0];
    const int*   ei  = (const int*)d_in[1];
    const float* W1  = (const float*)d_in[2];
    const float* b1  = (const float*)d_in[3];
    const float* W2  = (const float*)d_in[4];
    const float* b2  = (const float*)d_in[5];
    const float* Wo  = (const float*)d_in[6];
    const float* bo  = (const float*)d_in[7];
    float* out = (float*)d_out;

    const int H    = in_sizes[3];              // 128
    const int FIN  = in_sizes[2] / H;          // 500
    const int N    = in_sizes[0] / FIN;        // 50000
    const int E    = in_sizes[1] / 2;          // 800000
    const int Kpad1 = ((FIN + 31) / 32) * 32;  // 512

    const int* src = ei;
    const int* dst = ei + E;

    // workspace layout (256B aligned)
    char* ws = (char*)d_ws;
    size_t off = 0;
    auto carve = [&](size_t bytes) { char* p = ws + off; off += (bytes + 255) & ~(size_t)255; return p; };
    float*    dis  = (float*)   carve((size_t)N * 4);
    _Float16* W1p  = (_Float16*)carve((size_t)Kpad1 * 128 * 2);
    _Float16* W2p  = (_Float16*)carve((size_t)128 * 128 * 2);
    _Float16* xh   = (_Float16*)carve((size_t)N * Kpad1 * 2);
    float*    hbuf = (float*)   carve((size_t)N * 128 * 4);
    float*    agg  = (float*)   carve((size_t)N * 128 * 4);
    _Float16* hbf  = (_Float16*)carve((size_t)N * 128 * 2);

    const int T = 256;
    // --- normalization -----------------------------------------------------
    deg_init_kernel<<<(N + T - 1) / T, T, 0, stream>>>(dis, N);
    deg_edges_kernel<<<(E + T - 1) / T, T, 0, stream>>>(dst, dis, E);
    rsqrt_kernel<<<(N + T - 1) / T, T, 0, stream>>>(dis, N);

    // --- input conversion + weight packing ---------------------------------
    cvt_f16_pad_kernel<<<dim3((Kpad1 + T - 1) / T, N), T, 0, stream>>>(x, xh, FIN, Kpad1);
    pack_w_kernel<<<(Kpad1 * 128 + T - 1) / T, T, 0, stream>>>(W1, W1p, FIN, Kpad1);
    pack_w_kernel<<<(128 * 128 + T - 1) / T, T, 0, stream>>>(W2, W2p, 128, 128);

    const int gemm_blocks = N / 16;            // 3125

    // --- layer 1 -----------------------------------------------------------
    gemm_wmma_f16_kernel<<<gemm_blocks, T, 0, stream>>>(xh, Kpad1, Kpad1 / 32, W1p, hbuf);
    agg_init_kernel<<<(N * 32 + T - 1) / T, T, 0, stream>>>(hbuf, dis, agg, N);
    scatter_kernel<<<(E * 32 + T - 1) / T, T, 0, stream>>>(src, dst, dis, hbuf, agg, E);
    bias_relu_f16_kernel<<<(N * 128 + T - 1) / T, T, 0, stream>>>(agg, b1, hbf, N * 128);

    // --- layer 2 -----------------------------------------------------------
    gemm_wmma_f16_kernel<<<gemm_blocks, T, 0, stream>>>(hbf, 128, 128 / 32, W2p, hbuf);
    agg_init_kernel<<<(N * 32 + T - 1) / T, T, 0, stream>>>(hbuf, dis, agg, N);
    scatter_kernel<<<(E * 32 + T - 1) / T, T, 0, stream>>>(src, dst, dis, hbuf, agg, E);
    bias_relu_f16_kernel<<<(N * 128 + T - 1) / T, T, 0, stream>>>(agg, b2, hbf, N * 128);

    // --- output layer + softmax -------------------------------------------
    out_softmax_kernel<<<(N + 7) / 8, T, 0, stream>>>(hbf, Wo, bo, out, N);
}
